// Attn_45372034515281
// MI455X (gfx1250) — matrix operand order repaired and checked by reading, not compiled, not run on hardware
//
#include <hip/hip_runtime.h>
#include <hip/hip_bf16.h>

// Problem constants (match reference)
#define BB   2
#define NNv  2048
#define DDv  1024
#define HHv  16
#define DHv  64
#define TDv  (3 * DDv)
#define ROWS (BB * NNv)          // 4096

typedef __attribute__((ext_vector_type(16))) __bf16 v16bf;
typedef __attribute__((ext_vector_type(8)))  float  v8f;
typedef __attribute__((ext_vector_type(4)))  int    v4i;

union ABfrag { v16bf v; uint4 q[2]; };

__device__ __forceinline__ unsigned short f2bf(float f) {
    unsigned int u = __float_as_uint(f);
    u += 0x7FFFu + ((u >> 16) & 1u);          // round-to-nearest-even
    return (unsigned short)(u >> 16);
}

__device__ __forceinline__ v8f wmma_bf16(const ABfrag& a, const ABfrag& b, v8f c) {
    return __builtin_amdgcn_wmma_f32_16x16x32_bf16(
        /*neg_a=*/false, a.v, /*neg_b=*/false, b.v,
        /*c_mod=*/(short)0, c, /*reuse_a=*/false, /*reuse_b=*/false);
}

// -------- async global->LDS staging (ASYNCcnt path), guarded by __has_builtin
#if __has_builtin(__builtin_amdgcn_global_load_async_to_lds_b128)
#define USE_ASYNC_LDS 1
typedef __attribute__((address_space(1))) v4i g_v4i;
typedef __attribute__((address_space(3))) v4i l_v4i;
#else
#define USE_ASYNC_LDS 0
#endif

__device__ __forceinline__ void stage_b128(const unsigned short* gsrc,
                                           unsigned short* ldst) {
#if USE_ASYNC_LDS
    __builtin_amdgcn_global_load_async_to_lds_b128(
        (g_v4i*)(uintptr_t)gsrc,
        (l_v4i*)(unsigned)(uintptr_t)ldst,
        0, 0);
#else
    *(uint4*)ldst = *(const uint4*)gsrc;
#endif
}

__device__ __forceinline__ void stage_wait() {
#if USE_ASYNC_LDS
#if __has_builtin(__builtin_amdgcn_s_wait_asynccnt)
    __builtin_amdgcn_s_wait_asynccnt(0);
#else
    asm volatile("s_wait_asynccnt 0x0" ::: "memory");
#endif
#endif
}

// ---------------------------------------------------------------- convert
__global__ void cvt_kernel(const float* __restrict__ src,
                           unsigned short* __restrict__ dst, int n) {
    int i = blockIdx.x * blockDim.x + threadIdx.x;
    int stride = gridDim.x * blockDim.x;
    for (; i < n; i += stride) dst[i] = f2bf(src[i]);
}

__global__ void zero_kernel(float* __restrict__ p, int n) {
    int i = blockIdx.x * blockDim.x + threadIdx.x;
    if (i < n) p[i] = 0.0f;
}

// ---------------------------------------------------------------- bf16 GEMM
// C[M,NC] = A[M,K] * B[K,NC] (+bias). A,B bf16 row-major, C fp32.
// Block: 128 threads (4 waves). Tile: 64 rows x 64 cols, K-step 32.
// Wave w computes rows [w*16, w*16+16) x 64 cols = 4 accumulators.
template <bool BIAS>
__global__ void gemm_bf16_kernel(const unsigned short* __restrict__ A,
                                 const unsigned short* __restrict__ Bm,
                                 const float* __restrict__ bias,
                                 float* __restrict__ C,
                                 int M, int K, int NC) {
    __shared__ unsigned short As[64][40];   // [row][k], +8 pad
    __shared__ unsigned short Bs[64][40];   // transposed: [c][k], +8 pad

    const int row0 = blockIdx.y * 64;
    const int col0 = blockIdx.x * 64;
    const int tid  = threadIdx.x;
    const int lane = tid & 31, wv = tid >> 5;
    const int lr   = lane & 15, half = lane >> 4;

    // A staging: thread -> (row, 16-elem half) of 64x32 tile
    const int arow = tid >> 1, ah = (tid & 1) * 16;
    // B staging: thread -> (k-row, 16-col group) of 32x64 tile
    const int bk = tid >> 2, bj = (tid & 3) * 16;

    v8f acc[4] = {};

    for (int k0 = 0; k0 < K; k0 += 32) {
        // stage A tile 64x32 via async LDS DMA (2 x b128 per thread)
        const unsigned short* ga = &A[(size_t)(row0 + arow) * K + k0 + ah];
        stage_b128(ga,     &As[arow][ah]);
        stage_b128(ga + 8, &As[arow][ah + 8]);

        // stage B tile 32x64 transposed: coalesced b128 reads, b16 LDS writes
        uint4 br[2];
        br[0] = *(const uint4*)&Bm[(size_t)(k0 + bk) * NC + col0 + bj];
        br[1] = *(const uint4*)&Bm[(size_t)(k0 + bk) * NC + col0 + bj + 8];
        const unsigned short* bw = (const unsigned short*)br;
        #pragma unroll
        for (int i = 0; i < 16; ++i)
            Bs[bj + i][bk] = bw[i];

        stage_wait();            // wave's async A copies landed in LDS
        __syncthreads();         // all waves' stores visible

        ABfrag a, b;
        a.q[0] = *(const uint4*)&As[wv * 16 + lr][half * 16];
        a.q[1] = *(const uint4*)&As[wv * 16 + lr][half * 16 + 8];
        #pragma unroll
        for (int ct = 0; ct < 4; ++ct) {
            const unsigned short* bp = &Bs[ct * 16 + lr][half * 16];
            b.q[0] = *(const uint4*)bp;
            b.q[1] = *(const uint4*)(bp + 8);
            acc[ct] = wmma_bf16(a, b, acc[ct]);
        }
        __syncthreads();
    }

    const int rbase = row0 + wv * 16 + 8 * half;
    #pragma unroll
    for (int ct = 0; ct < 4; ++ct) {
        const int col = col0 + ct * 16 + lr;
        const float bv = BIAS ? bias[col] : 0.0f;
        #pragma unroll
        for (int r = 0; r < 8; ++r)
            C[(size_t)(rbase + r) * NC + col] = acc[ct][r] + bv;
    }
}

// ---------------------------------------------------------------- norm+rope
// One wave handles one (q|k, b, h, n) row of dh=64: RMSNorm * scale, RoPE,
// emit bf16 into head-major Qh/Kh [b][h][n][dh].
__global__ void normrope_kernel(const float* __restrict__ QKV,
                                const float* __restrict__ scale_param,
                                unsigned short* __restrict__ Qh,
                                unsigned short* __restrict__ Kh) {
    const int wv = threadIdx.x >> 5, lane = threadIdx.x & 31;
    const int w  = blockIdx.x * 8 + wv;     // 0 .. 2*B*H*N-1
    const int qk = w >> 16;                 // B*H*N = 65536
    const int r  = w & 65535;
    const int b  = r >> 15;                 // H*N = 32768
    const int hn = r & 32767;
    const int h  = hn >> 11;                // N = 2048
    const int n  = hn & 2047;

    const float* src = QKV + (size_t)(b * NNv + n) * TDv + qk * DDv + h * DHv;
    const int c1 = lane, c2 = lane + 32;
    float t1 = src[c1], t2 = src[c2];

    float ss = t1 * t1 + t2 * t2;
    #pragma unroll
    for (int off = 16; off >= 1; off >>= 1) ss += __shfl_xor(ss, off, 32);
    const float rinv = rsqrtf(ss * (1.0f / 64.0f) + 1e-6f);

    const float* sp = scale_param + h * DHv;
    t1 = t1 * rinv * (sp[c1] * 32.0f);      // scaler = scale_param * sqrt(D)
    t2 = t2 * rinv * (sp[c2] * 32.0f);

    // inv_freq[c1] = 10000^(-c1/32) ; ang = n * inv_freq
    const float invf = __expf(-(float)c1 * 0.28782313662425572f); // ln(1e4)/32
    const float ang  = (float)n * invf;
    float sn, cs;
    __sincosf(ang, &sn, &cs);

    unsigned short* dst = (qk ? Kh : Qh) + (size_t)((b * HHv + h) * NNv + n) * DHv;
    dst[c1] = f2bf(t1 * cs - t2 * sn);
    dst[c2] = f2bf(t1 * sn + t2 * cs);
}

// ---------------------------------------------------------------- attention
// Per block: one (b,h) pair and a 16-row n-tile. Two-phase flash-style pass:
// phase 1 computes row max/sumexp online, phase 2 recomputes scores and
// accumulates softmax column sums into colw[b][h][m] via fp32 atomics.
__global__ void attn_colsum_kernel(const unsigned short* __restrict__ Qh,
                                   const unsigned short* __restrict__ Kh,
                                   float* __restrict__ colw) {
    const int tid  = threadIdx.x;
    const int lane = tid & 31, wv = tid >> 5;     // 8 waves
    const int lr   = lane & 15, half = lane >> 4;
    const int nt   = blockIdx.x & 127;            // N/16 tiles
    const int bh   = blockIdx.x >> 7;

    const unsigned short* Qb  = Qh + (size_t)(bh * NNv + nt * 16) * DHv;
    const unsigned short* Kb0 = Kh + (size_t)bh * NNv * DHv;

    // A fragments (Q 16x64 = two K-steps), contiguous k = 16*half + s (matches B)
    ABfrag a0, a1;
    {
        const unsigned short* qr = Qb + lr * DHv;
        a0.q[0] = *(const uint4*)(qr + half * 16);
        a0.q[1] = *(const uint4*)(qr + half * 16 + 8);
        a1.q[0] = *(const uint4*)(qr + 32 + half * 16);
        a1.q[1] = *(const uint4*)(qr + 32 + half * 16 + 8);
    }

    float rm[8], rl[8];
    #pragma unroll
    for (int r = 0; r < 8; ++r) { rm[r] = -3.0e38f; rl[r] = 0.0f; }

    // ---- phase 1: online row max / sumexp over all m tiles
    for (int ct = wv; ct < 128; ct += 8) {
        const unsigned short* kr = Kb0 + (size_t)(ct * 16 + lr) * DHv;
        ABfrag b0, b1;
        b0.q[0] = *(const uint4*)(kr + half * 16);
        b0.q[1] = *(const uint4*)(kr + half * 16 + 8);
        b1.q[0] = *(const uint4*)(kr + 32 + half * 16);
        b1.q[1] = *(const uint4*)(kr + 32 + half * 16 + 8);
        v8f acc = {};
        acc = wmma_bf16(a0, b0, acc);
        acc = wmma_bf16(a1, b1, acc);
        #pragma unroll
        for (int r = 0; r < 8; ++r) {
            float s  = acc[r] * 8.0f;                 // * sqrt(DH)
            float mn = fmaxf(rm[r], s);
            rl[r] = rl[r] * __expf(rm[r] - mn) + __expf(s - mn);
            rm[r] = mn;
        }
    }
    // merge stats across the 16 lanes of each half (columns)
    #pragma unroll
    for (int off = 1; off <= 8; off <<= 1) {
        #pragma unroll
        for (int r = 0; r < 8; ++r) {
            float om = __shfl_xor(rm[r], off, 32);
            float ol = __shfl_xor(rl[r], off, 32);
            float mn = fmaxf(rm[r], om);
            rl[r] = rl[r] * __expf(rm[r] - mn) + ol * __expf(om - mn);
            rm[r] = mn;
        }
    }

    __shared__ float pmax[8][16], psum[8][16];
    __shared__ float rowmax[16], rowsinv[16];
    if (lr == 0) {
        #pragma unroll
        for (int r = 0; r < 8; ++r) {
            pmax[wv][r + 8 * half] = rm[r];
            psum[wv][r + 8 * half] = rl[r];
        }
    }
    __syncthreads();
    if (tid < 16) {
        float m = -3.0e38f, l = 0.0f;
        #pragma unroll
        for (int w = 0; w < 8; ++w) {
            float om = pmax[w][tid], ol = psum[w][tid];
            float mn = fmaxf(m, om);
            l = l * __expf(m - mn) + ol * __expf(om - mn);
            m = mn;
        }
        rowmax[tid]  = m;
        rowsinv[tid] = 1.0f / l;
    }
    __syncthreads();

    float rmx[8], rsi[8];
    #pragma unroll
    for (int r = 0; r < 8; ++r) {
        rmx[r] = rowmax[r + 8 * half];
        rsi[r] = rowsinv[r + 8 * half];
    }

    // ---- phase 2: recompute scores, accumulate softmax column sums
    for (int ct = wv; ct < 128; ct += 8) {
        const unsigned short* kr = Kb0 + (size_t)(ct * 16 + lr) * DHv;
        ABfrag b0, b1;
        b0.q[0] = *(const uint4*)(kr + half * 16);
        b0.q[1] = *(const uint4*)(kr + half * 16 + 8);
        b1.q[0] = *(const uint4*)(kr + 32 + half * 16);
        b1.q[1] = *(const uint4*)(kr + 32 + half * 16 + 8);
        v8f acc = {};
        acc = wmma_bf16(a0, b0, acc);
        acc = wmma_bf16(a1, b1, acc);
        float cs = 0.0f;
        #pragma unroll
        for (int r = 0; r < 8; ++r)
            cs += __expf(acc[r] * 8.0f - rmx[r]) * rsi[r];
        cs += __shfl_xor(cs, 16, 32);   // combine rows 0-7 and 8-15 (same col)
        if (half == 0)
            atomicAdd(&colw[(size_t)bh * NNv + ct * 16 + lr], cs);
    }
}

// ---------------------------------------------------------------- v * colsum
__global__ void vmul_kernel(const float* __restrict__ QKV,
                            const float* __restrict__ colw,
                            unsigned short* __restrict__ AOb) {
    const int i   = blockIdx.x * blockDim.x + threadIdx.x;  // over ROWS*D
    const int col = i & (DDv - 1);
    const int row = i >> 10;            // b*N + m
    const int b   = row >> 11;          // N = 2048
    const int m   = row & 2047;
    const int h   = col >> 6;
    const float v = QKV[(size_t)row * TDv + 2 * DDv + col];
    const float w = colw[(size_t)(b * HHv + h) * NNv + m];
    AOb[i] = f2bf(v * w);
}

// ---------------------------------------------------------------- launch
extern "C" void kernel_launch(void* const* d_in, const int* in_sizes, int n_in,
                              void* d_out, int out_size, void* d_ws, size_t ws_size,
                              hipStream_t stream) {
    const float* x           = (const float*)d_in[0];
    const float* w_qkv       = (const float*)d_in[1];
    const float* w_out       = (const float*)d_in[2];
    const float* b_out       = (const float*)d_in[3];
    const float* scale_param = (const float*)d_in[4];
    float* out = (float*)d_out;

    char* ws = (char*)d_ws;
    auto alloc = [&](size_t bytes) {
        char* p = ws;
        ws += (bytes + 255) & ~(size_t)255;
        return p;
    };
    unsigned short* Xb    = (unsigned short*)alloc((size_t)ROWS * DDv * 2);
    unsigned short* Wqkvb = (unsigned short*)alloc((size_t)DDv * TDv * 2);
    unsigned short* Woutb = (unsigned short*)alloc((size_t)DDv * DDv * 2);
    float*          QKV   = (float*)         alloc((size_t)ROWS * TDv * 4);
    unsigned short* Qh    = (unsigned short*)alloc((size_t)BB * HHv * NNv * DHv * 2);
    unsigned short* Kh    = (unsigned short*)alloc((size_t)BB * HHv * NNv * DHv * 2);
    float*          colw  = (float*)         alloc((size_t)BB * HHv * NNv * 4);
    unsigned short* AOb   = (unsigned short*)alloc((size_t)ROWS * DDv * 2);

    cvt_kernel<<<1024, 256, 0, stream>>>(x, Xb, ROWS * DDv);
    cvt_kernel<<<1024, 256, 0, stream>>>(w_qkv, Wqkvb, DDv * TDv);
    cvt_kernel<<<1024, 256, 0, stream>>>(w_out, Woutb, DDv * DDv);
    zero_kernel<<<(BB * HHv * NNv) / 256, 256, 0, stream>>>(colw, BB * HHv * NNv);

    // QKV = X @ Wqkv   (64x64 tiles)
    gemm_bf16_kernel<false><<<dim3(TDv / 64, ROWS / 64), 128, 0, stream>>>(
        Xb, Wqkvb, nullptr, QKV, ROWS, DDv, TDv);

    // RMSNorm * scale + RoPE, head-major bf16 Q/K  (one wave per row)
    normrope_kernel<<<(2 * BB * HHv * NNv) / 8, 256, 0, stream>>>(
        QKV, scale_param, Qh, Kh);

    // softmax column sums
    attn_colsum_kernel<<<BB * HHv * (NNv / 16), 256, 0, stream>>>(Qh, Kh, colw);

    // attn_out = v * colsum (bf16)
    vmul_kernel<<<(ROWS * DDv) / 256, 256, 0, stream>>>(QKV, colw, AOb);

    // out = attn_out @ w_out + b_out
    gemm_bf16_kernel<true><<<dim3(DDv / 64, ROWS / 64), 128, 0, stream>>>(
        AOb, Woutb, b_out, out, ROWS, DDv, DDv);
}
